// PointNet2_15152644620608
// MI455X (gfx1250) — matrix operand order repaired and checked by reading, not compiled, not run on hardware
//
#include <hip/hip_runtime.h>
#include <hip/hip_fp16.h>

// ---------------------------------------------------------------------------
// PointNet++ forward for MI455X (gfx1250, wave32).
// GEMMs (the FLOP-dominant part) run on V_WMMA_F32_16X16X32_F16:
//   f32 activations/weights -> f16 fragments via LDS staging, f32 accum.
// Shapes in this net: all GEMM M are multiples of 64, all N multiples of 32;
// only K is ragged (3/131/259/515) -> one zero-padded tail tile whose loads
// are index-clamped (unconditional) with a post-load select. Staging is
// two-phase (batch global loads into temps, then batch LDS stores) so the
// compiler clauses the loads instead of chaining load->wait per element.
// Both A and B fragments are read from LDS as two ds_load_b128 each: A is
// K-contiguous natively; B is stored column-major in LDS to make it so.
// LDS rows padded to 40 halves (80B: 16B-aligned, bank-conflict friendly).
// ---------------------------------------------------------------------------

typedef __attribute__((ext_vector_type(16))) _Float16 v16h;
typedef __attribute__((ext_vector_type(8)))  float    v8f;

#define BN_EPS 1e-5f
#define LDS_STRIDE 40  // halves per row/column slot: 80B, 16B-aligned

// ------------------------------ GEMM (WMMA) --------------------------------
// Y[M,N] = X[M,K] * W[K,N] + bias[N]
// Block = 256 threads = 8 waves arranged 4x2; block tile = 64 rows x 32 cols.
// Each wave owns one 16x16 WMMA tile.
__global__ __launch_bounds__(256)
void pn2_gemm_bias_wmma(const float* __restrict__ X, const float* __restrict__ W,
                        const float* __restrict__ bias, float* __restrict__ Y,
                        int M, int N, int K) {
    __shared__ _Float16 As[64 * LDS_STRIDE];   // A tile, row-major [row][k]
    __shared__ _Float16 Bs[32 * LDS_STRIDE];   // B tile, col-major [n][k]

    const int tid  = threadIdx.x;
    const int lane = tid & 31;
    const int wave = tid >> 5;
    const int wm   = wave >> 1;        // 0..3 : 16-row group
    const int wn   = wave & 1;         // 0..1 : 16-col group
    const int row0 = blockIdx.x * 64;
    const int col0 = blockIdx.y * 32;
    const int half = lane >> 4;        // lanes 0-15 vs 16-31
    const int l15  = lane & 15;

    v8f acc = {};
    const int Kfull = K & ~31;         // last multiple-of-32 boundary

    for (int kb = 0; kb < K; kb += 32) {
        float av[8], bv4[4];
        if (kb < Kfull) {
            // ---- full tile: batch all loads, then batch all LDS stores ------
#pragma unroll
            for (int i = 0; i < 8; ++i) {
                int e = tid + i * 256;
                int r = e >> 5, c = e & 31;
                av[i] = X[(size_t)(row0 + r) * K + kb + c];
            }
#pragma unroll
            for (int i = 0; i < 4; ++i) {
                int e = tid + i * 256;
                int r = e >> 5, c = e & 31;
                bv4[i] = W[(size_t)(kb + r) * N + col0 + c];
            }
#pragma unroll
            for (int i = 0; i < 8; ++i) {
                int e = tid + i * 256;
                int r = e >> 5, c = e & 31;
                As[r * LDS_STRIDE + c] = (_Float16)av[i];
            }
#pragma unroll
            for (int i = 0; i < 4; ++i) {
                int e = tid + i * 256;
                int r = e >> 5, c = e & 31;
                Bs[c * LDS_STRIDE + r] = (_Float16)bv4[i];
            }
            // prefetch next A tile (lowers to global_prefetch_b8)
            if (kb + 32 < Kfull) {
                int r = tid >> 5, c = tid & 31;
                __builtin_prefetch(&X[(size_t)(row0 + r) * K + kb + 32 + c], 0, 0);
            }
        } else {
            // ---- single ragged tail tile: zero-pad K ------------------------
            // Clamp indices so every load is in-bounds & unconditional; the
            // padding is a post-load select (v_cndmask), not a masked load.
#pragma unroll
            for (int i = 0; i < 8; ++i) {
                int e = tid + i * 256;
                int r = e >> 5, c = e & 31;
                int kc = kb + c;
                int kcl = kc < K ? kc : K - 1;
                av[i] = X[(size_t)(row0 + r) * K + kcl];
            }
#pragma unroll
            for (int i = 0; i < 4; ++i) {
                int e = tid + i * 256;
                int r = e >> 5, c = e & 31;
                int kr = kb + r;
                int krl = kr < K ? kr : K - 1;
                bv4[i] = W[(size_t)krl * N + col0 + c];
            }
#pragma unroll
            for (int i = 0; i < 8; ++i) {
                int e = tid + i * 256;
                int r = e >> 5, c = e & 31;
                As[r * LDS_STRIDE + c] =
                    (_Float16)((kb + c) < K ? av[i] : 0.f);
            }
#pragma unroll
            for (int i = 0; i < 4; ++i) {
                int e = tid + i * 256;
                int r = e >> 5, c = e & 31;
                Bs[c * LDS_STRIDE + r] =
                    (_Float16)((kb + r) < K ? bv4[i] : 0.f);
            }
        }
        __syncthreads();

        // A fragment (ISA 7.12.2, 16-bit A 16x32): lane owns row (lane&15);
        // halves K = {half*8 + 0..7} then {16 + half*8 + 0..7} -> 2x b128.
        v16h a;
        {
            const uint4* Ap4 =
                (const uint4*)(As + (size_t)(wm * 16 + l15) * LDS_STRIDE);
            ((uint4*)&a)[0] = Ap4[half];       // dwords half*4 .. half*4+3
            ((uint4*)&a)[1] = Ap4[2 + half];   // dwords 8+half*4 .. +3
        }

        // B fragment: lane owns column; K = {half*8 + 0..7} then
        // {16 + half*8 + 0..7} -> 2x b128 from the column-major tile.
        v16h bf;
        {
            const uint4* Bp4 = (const uint4*)Bs;
            int cb = (wn * 16 + l15) * (LDS_STRIDE / 8) + half; // uint4 idx
            ((uint4*)&bf)[0] = Bp4[cb];
            ((uint4*)&bf)[1] = Bp4[cb + 2];
        }

        acc = __builtin_amdgcn_wmma_f32_16x16x32_f16(
            /*neg_a=*/false, a, /*neg_b=*/false, bf,
            /*c_mod=*/(short)0, acc, /*reuse_a=*/false, /*reuse_b=*/false);
        __syncthreads();
    }

    // C/D layout: VGPR i -> row = i (+8 for lanes 16-31), col = lane&15.
    // M % 64 == 0 and N % 32 == 0 for every layer -> unconditional stores.
    const int col = col0 + wn * 16 + l15;
    const float bv = bias[col];
    float* yp = Y + (size_t)(row0 + wm * 16 + half * 8) * N + col;
#pragma unroll
    for (int i = 0; i < 8; ++i)
        yp[(size_t)i * N] = acc[i] + bv;
}

// ------------------------- BatchNorm (batch stats) -------------------------
// One block per channel: reduce mean/var over M rows, emit mean & rsqrt(var).
__global__ __launch_bounds__(256)
void pn2_bn_stats(const float* __restrict__ Y, int M, int N,
                  float* __restrict__ mr) {
    const int c = blockIdx.x;
    float s = 0.f, ss = 0.f;
    for (int r = threadIdx.x; r < M; r += 256) {
        float v = Y[(size_t)r * N + c];
        s += v; ss += v * v;
    }
    __shared__ float sh[256], sh2[256];
    sh[threadIdx.x] = s; sh2[threadIdx.x] = ss;
    __syncthreads();
    for (int st = 128; st > 0; st >>= 1) {
        if (threadIdx.x < st) {
            sh[threadIdx.x]  += sh[threadIdx.x + st];
            sh2[threadIdx.x] += sh2[threadIdx.x + st];
        }
        __syncthreads();
    }
    if (threadIdx.x == 0) {
        float mean = sh[0] / (float)M;
        float var  = sh2[0] / (float)M - mean * mean;
        mr[2 * c]     = mean;
        mr[2 * c + 1] = rsqrtf(var + BN_EPS);
    }
}

__global__ __launch_bounds__(256)
void pn2_bn_apply_relu(float* __restrict__ Y, long long total, int N,
                       const float* __restrict__ mr,
                       const float* __restrict__ gamma,
                       const float* __restrict__ beta) {
    long long i = (long long)blockIdx.x * 256 + threadIdx.x;
    if (i >= total) return;
    int c = (int)(i % N);
    float v = (Y[i] - mr[2 * c]) * mr[2 * c + 1] * gamma[c] + beta[c];
    Y[i] = v > 0.f ? v : 0.f;
}

// ------------------------------- FPS ---------------------------------------
// One block per batch; running min-distance array lives in LDS (<=32KB).
__global__ __launch_bounds__(256)
void pn2_fps(const float* __restrict__ xyz, int n, int npoint,
             int* __restrict__ out) {
    const int b = blockIdx.x;
    const float* __restrict__ p = xyz + (size_t)b * n * 3;
    __shared__ float dist[8192];
    __shared__ float rv[256];
    __shared__ int   ri[256];
    const int t = threadIdx.x;
    for (int j = t; j < n; j += 256) dist[j] = 1e10f;
    __syncthreads();
    int cur = 0;
    for (int it = 0; it < npoint; ++it) {
        if (t == 0) out[b * npoint + it] = cur;
        float cx = p[cur * 3], cy = p[cur * 3 + 1], cz = p[cur * 3 + 2];
        float bestv = -1.f; int besti = 0;
        for (int j = t; j < n; j += 256) {
            float dx = p[j * 3] - cx, dy = p[j * 3 + 1] - cy, dz = p[j * 3 + 2] - cz;
            float d = dx * dx + dy * dy + dz * dz;
            float m = fminf(dist[j], d);
            dist[j] = m;
            if (m > bestv) { bestv = m; besti = j; }
        }
        rv[t] = bestv; ri[t] = besti;
        __syncthreads();
        for (int st = 128; st > 0; st >>= 1) {
            if (t < st && rv[t + st] > rv[t]) { rv[t] = rv[t + st]; ri[t] = ri[t + st]; }
            __syncthreads();
        }
        cur = ri[0];
        __syncthreads();
    }
}

// --------------------------- gather / ball query ----------------------------
__global__ void pn2_gather_xyz(const float* __restrict__ xyz, int n,
                               const int* __restrict__ idx, int total,
                               int S, float* __restrict__ out) {
    int i = blockIdx.x * blockDim.x + threadIdx.x;
    if (i >= total) return;
    int b = i / S;
    int j = idx[i];
    const float* p = xyz + ((size_t)b * n + j) * 3;
    out[i * 3]     = p[0];
    out[i * 3 + 1] = p[1];
    out[i * 3 + 2] = p[2];
}

// torch query_ball_point semantics: first K in-radius indices in ascending
// order; leftovers take the first valid index.
__global__ void pn2_ball_query(float r2, int Ks, const float* __restrict__ xyz,
                               int n, const float* __restrict__ nxyz,
                               int total, int S, int* __restrict__ gidx) {
    int i = blockIdx.x * blockDim.x + threadIdx.x;
    if (i >= total) return;
    int b = i / S;
    const float* p = xyz + (size_t)b * n * 3;
    float cx = nxyz[i * 3], cy = nxyz[i * 3 + 1], cz = nxyz[i * 3 + 2];
    int* out = gidx + (size_t)i * Ks;
    int cnt = 0, first = -1;
    for (int j = 0; j < n && cnt < Ks; ++j) {
        float dx = p[j * 3] - cx, dy = p[j * 3 + 1] - cy, dz = p[j * 3 + 2] - cz;
        if (dx * dx + dy * dy + dz * dz <= r2) {
            if (first < 0) first = j;
            out[cnt++] = j;
        }
    }
    if (first < 0) first = 0;
    for (; cnt < Ks; ++cnt) out[cnt] = first;
}

// Build MLP input rows: [rel_xyz | gathered features]
__global__ void pn2_group(const float* __restrict__ xyz,
                          const float* __restrict__ pts, int Cin,
                          const int* __restrict__ gidx,
                          const float* __restrict__ nxyz,
                          int n, int S, int Ks, long long total,
                          float* __restrict__ Xo) {
    long long i = (long long)blockIdx.x * blockDim.x + threadIdx.x;
    if (i >= total) return;
    int bs = (int)(i / Ks);
    int b = bs / S;
    int j = gidx[i];
    const float* p = xyz + ((size_t)b * n + j) * 3;
    int ld = 3 + Cin;
    float* row = Xo + (size_t)i * ld;
    row[0] = p[0] - nxyz[bs * 3];
    row[1] = p[1] - nxyz[bs * 3 + 1];
    row[2] = p[2] - nxyz[bs * 3 + 2];
    if (pts) {
        const float* q = pts + ((size_t)b * n + j) * Cin;
        for (int c = 0; c < Cin; ++c) row[3 + c] = q[c];
    }
}

// ----------------------------- max pool ------------------------------------
__global__ void pn2_maxpool(const float* __restrict__ Y, long long total,
                            int Ks, int C, float* __restrict__ out) {
    long long i = (long long)blockIdx.x * blockDim.x + threadIdx.x;
    if (i >= total) return;
    long long r = i / C;
    int c = (int)(i % C);
    const float* p = Y + (size_t)r * Ks * C + c;
    float m = -1e30f;
    for (int k = 0; k < Ks; ++k) m = fmaxf(m, p[(size_t)k * C]);
    out[i] = m;
}

// ----------------------- concat / broadcast / copy --------------------------
__global__ void pn2_copy_cols(const float* __restrict__ A, int Ca,
                              float* __restrict__ X, int ld, int coloff,
                              long long total) {
    long long i = (long long)blockIdx.x * blockDim.x + threadIdx.x;
    if (i >= total) return;
    long long r = i / Ca;
    int c = (int)(i % Ca);
    X[r * ld + coloff + c] = A[i];
}

__global__ void pn2_bcast_cols(const float* __restrict__ src, int C,
                               float* __restrict__ X, int ld, int coloff,
                               int nper, long long total) {
    long long i = (long long)blockIdx.x * blockDim.x + threadIdx.x;
    if (i >= total) return;
    int c = (int)(i % C);
    long long r = i / C;
    int b = (int)(r / nper);
    X[r * ld + coloff + c] = src[(size_t)b * C + c];
}

__global__ void pn2_copy_flat(const float* __restrict__ A,
                              float* __restrict__ Bp, long long n) {
    long long i = (long long)blockIdx.x * blockDim.x + threadIdx.x;
    if (i < n) Bp[i] = A[i];
}

// ------------------------- 3-NN interpolation -------------------------------
__global__ void pn2_interp3(const float* __restrict__ x1, int n1,
                            const float* __restrict__ x2, int n2,
                            const float* __restrict__ p2, int C,
                            float* __restrict__ X, int ld, int coloff,
                            int total) {
    int i = blockIdx.x * blockDim.x + threadIdx.x;
    if (i >= total) return;
    int b = i / n1;
    float cx = x1[i * 3], cy = x1[i * 3 + 1], cz = x1[i * 3 + 2];
    const float* q = x2 + (size_t)b * n2 * 3;
    float d0 = 1e30f, d1 = 1e30f, d2 = 1e30f;
    int i0 = 0, i1 = 0, i2 = 0;
    for (int j = 0; j < n2; ++j) {
        float dx = q[j * 3] - cx, dy = q[j * 3 + 1] - cy, dz = q[j * 3 + 2] - cz;
        float d = dx * dx + dy * dy + dz * dz;
        if (d < d0)      { d2 = d1; i2 = i1; d1 = d0; i1 = i0; d0 = d; i0 = j; }
        else if (d < d1) { d2 = d1; i2 = i1; d1 = d;  i1 = j; }
        else if (d < d2) { d2 = d;  i2 = j; }
    }
    float w0 = 1.f / (d0 + 1e-8f), w1 = 1.f / (d1 + 1e-8f), w2 = 1.f / (d2 + 1e-8f);
    float ws = w0 + w1 + w2;
    w0 /= ws; w1 /= ws; w2 /= ws;
    const float* f0 = p2 + ((size_t)b * n2 + i0) * C;
    const float* f1 = p2 + ((size_t)b * n2 + i1) * C;
    const float* f2 = p2 + ((size_t)b * n2 + i2) * C;
    float* row = X + (size_t)i * ld + coloff;
    for (int c = 0; c < C; ++c)
        row[c] = w0 * f0[c] + w1 * f1[c] + w2 * f2[c];
}

// --------------------------- final transpose --------------------------------
// [B,N,C] -> [B,C,N]
__global__ void pn2_transpose_out(const float* __restrict__ Yin, int n, int C,
                                  long long total, float* __restrict__ out) {
    long long i = (long long)blockIdx.x * blockDim.x + threadIdx.x;
    if (i >= total) return;
    int c = (int)(i % C);
    long long t = i / C;
    int j = (int)(t % n);
    int b = (int)(t / n);
    out[((size_t)b * C + c) * n + j] = Yin[i];
}

// ===========================================================================
//                                 host side
// ===========================================================================
struct PN2Layer { const float *W, *b, *g, *be; int Cin, Cout; };

extern "C" void kernel_launch(void* const* d_in, const int* in_sizes, int n_in,
                              void* d_out, int out_size, void* d_ws, size_t ws_size,
                              hipStream_t stream) {
    (void)in_sizes; (void)n_in; (void)out_size; (void)ws_size;
    const int B = 8, N = 8192;
    const float* xyz = (const float*)d_in[0];

    // ---- parse params (insertion order: sa1..sa4, fp4..fp1; W,b,gamma,beta)
    int pi = 1;
    auto mk = [&](int cin, const int* couts, int nl, PN2Layer* L) {
        for (int i = 0; i < nl; ++i) {
            L[i].W  = (const float*)d_in[pi++];
            L[i].b  = (const float*)d_in[pi++];
            L[i].g  = (const float*)d_in[pi++];
            L[i].be = (const float*)d_in[pi++];
            L[i].Cin = cin; L[i].Cout = couts[i]; cin = couts[i];
        }
    };
    PN2Layer sa1[3], sa2[3], sa3[3], sa4[3], fp4[2], fp3[2], fp2[2], fp1[3];
    { int c[3] = {64, 64, 128};     mk(3,    c, 3, sa1); }
    { int c[3] = {128, 128, 256};   mk(131,  c, 3, sa2); }
    { int c[3] = {256, 256, 512};   mk(259,  c, 3, sa3); }
    { int c[3] = {512, 512, 1024};  mk(515,  c, 3, sa4); }
    { int c[2] = {512, 512};        mk(1536, c, 2, fp4); }
    { int c[2] = {512, 256};        mk(768,  c, 2, fp3); }
    { int c[2] = {256, 128};        mk(384,  c, 2, fp2); }
    { int c[3] = {128, 128, 128};   mk(128,  c, 3, fp1); }

    // ---- workspace bump allocator (deterministic layout every call)
    char* wp = (char*)d_ws;
    auto alloc = [&](size_t bytes) -> void* {
        void* r = (void*)wp;
        wp += (bytes + 255) & ~(size_t)255;
        return r;
    };
    float* stats = (float*)alloc(2 * 1024 * sizeof(float));
    int*   fidx1 = (int*)alloc((size_t)B * 1024 * 4);
    int*   fidx2 = (int*)alloc((size_t)B * 384 * 4);
    int*   fidx3 = (int*)alloc((size_t)B * 128 * 4);
    float* l1x = (float*)alloc((size_t)B * 1024 * 3 * 4);
    float* l2x = (float*)alloc((size_t)B * 384 * 3 * 4);
    float* l3x = (float*)alloc((size_t)B * 128 * 3 * 4);
    int*   gidx1 = (int*)alloc((size_t)B * 1024 * 64 * 4);
    int*   gidx2 = (int*)alloc((size_t)B * 384 * 64 * 4);
    int*   gidx3 = (int*)alloc((size_t)B * 128 * 64 * 4);
    float* l1p  = (float*)alloc((size_t)B * 1024 * 128 * 4);
    float* l2p  = (float*)alloc((size_t)B * 384 * 256 * 4);
    float* l3p  = (float*)alloc((size_t)B * 128 * 512 * 4);
    float* l4p  = (float*)alloc((size_t)B * 1024 * 4);
    float* l3p2 = (float*)alloc((size_t)B * 128 * 512 * 4);
    float* l2p2 = (float*)alloc((size_t)B * 384 * 256 * 4);
    float* l1p2 = (float*)alloc((size_t)B * 1024 * 128 * 4);
    const long long R1 = (long long)B * 1024 * 64;   // 524288
    const long long R2 = (long long)B * 384 * 64;    // 196608
    const long long R3 = (long long)B * 128 * 64;    // 65536
    float* bufA = (float*)alloc((size_t)R1 * 128 * 4);  // largest odd-layer out
    float* bufB = (float*)alloc((size_t)R1 * 64 * 4);   // largest even-layer out
    float* bufC = (float*)alloc((size_t)R2 * 131 * 4);  // largest staged input

    auto cdiv = [](long long a, long long b) { return (int)((a + b - 1) / b); };

    // ---- MLP runner: GEMM(WMMA) -> BN stats -> BN+ReLU, ping-pong bufA/bufB
    auto run_mlp = [&](const float* X0, int M, const PN2Layer* L, int nl) -> float* {
        const float* X = X0;
        float* Y = nullptr;
        for (int i = 0; i < nl; ++i) {
            Y = (i & 1) ? bufB : bufA;
            dim3 g(cdiv(M, 64), L[i].Cout / 32);
            pn2_gemm_bias_wmma<<<g, 256, 0, stream>>>(X, L[i].W, L[i].b, Y,
                                                      M, L[i].Cout, L[i].Cin);
            pn2_bn_stats<<<L[i].Cout, 256, 0, stream>>>(Y, M, L[i].Cout, stats);
            long long tot = (long long)M * L[i].Cout;
            pn2_bn_apply_relu<<<cdiv(tot, 256), 256, 0, stream>>>(
                Y, tot, L[i].Cout, stats, L[i].g, L[i].be);
            X = Y;
        }
        return Y;
    };

    // ================= SA1: 8192 -> 1024 pts, r=0.1, K=64 ==================
    pn2_fps<<<B, 256, 0, stream>>>(xyz, N, 1024, fidx1);
    pn2_gather_xyz<<<cdiv(B * 1024, 256), 256, 0, stream>>>(xyz, N, fidx1, B * 1024, 1024, l1x);
    pn2_ball_query<<<cdiv(B * 1024, 256), 256, 0, stream>>>(0.01f, 64, xyz, N, l1x, B * 1024, 1024, gidx1);
    pn2_group<<<cdiv(R1, 256), 256, 0, stream>>>(xyz, nullptr, 0, gidx1, l1x, N, 1024, 64, R1, bufC);
    {
        float* y = run_mlp(bufC, (int)R1, sa1, 3); // -> bufA [R1,128]
        pn2_maxpool<<<cdiv((long long)B * 1024 * 128, 256), 256, 0, stream>>>(
            y, (long long)B * 1024 * 128, 64, 128, l1p);
    }

    // ================= SA2: 1024 -> 384 pts, r=0.2, K=64 ===================
    pn2_fps<<<B, 256, 0, stream>>>(l1x, 1024, 384, fidx2);
    pn2_gather_xyz<<<cdiv(B * 384, 256), 256, 0, stream>>>(l1x, 1024, fidx2, B * 384, 384, l2x);
    pn2_ball_query<<<cdiv(B * 384, 256), 256, 0, stream>>>(0.04f, 64, l1x, 1024, l2x, B * 384, 384, gidx2);
    pn2_group<<<cdiv(R2, 256), 256, 0, stream>>>(l1x, l1p, 128, gidx2, l2x, 1024, 384, 64, R2, bufC);
    {
        float* y = run_mlp(bufC, (int)R2, sa2, 3); // -> bufA [R2,256]
        pn2_maxpool<<<cdiv((long long)B * 384 * 256, 256), 256, 0, stream>>>(
            y, (long long)B * 384 * 256, 64, 256, l2p);
    }

    // ================= SA3: 384 -> 128 pts, r=0.4, K=64 ====================
    pn2_fps<<<B, 256, 0, stream>>>(l2x, 384, 128, fidx3);
    pn2_gather_xyz<<<cdiv(B * 128, 256), 256, 0, stream>>>(l2x, 384, fidx3, B * 128, 128, l3x);
    pn2_ball_query<<<cdiv(B * 128, 256), 256, 0, stream>>>(0.16f, 64, l2x, 384, l3x, B * 128, 128, gidx3);
    pn2_group<<<cdiv(R3, 256), 256, 0, stream>>>(l2x, l2p, 256, gidx3, l3x, 384, 128, 64, R3, bufC);
    {
        float* y = run_mlp(bufC, (int)R3, sa3, 3); // -> bufA [R3,512]
        pn2_maxpool<<<cdiv((long long)B * 128 * 512, 256), 256, 0, stream>>>(
            y, (long long)B * 128 * 512, 64, 512, l3p);
    }

    // ================= SA4 (group-all): [B,128, 3+512] -> [B,1024] =========
    {
        long long M4 = (long long)B * 128;
        pn2_copy_cols<<<cdiv(M4 * 3, 256), 256, 0, stream>>>(l3x, 3, bufC, 515, 0, M4 * 3);
        pn2_copy_cols<<<cdiv(M4 * 512, 256), 256, 0, stream>>>(l3p, 512, bufC, 515, 3, M4 * 512);
        float* y = run_mlp(bufC, (int)M4, sa4, 3); // -> bufA [1024,1024]
        pn2_maxpool<<<cdiv((long long)B * 1024, 256), 256, 0, stream>>>(
            y, (long long)B * 1024, 128, 1024, l4p);
    }

    // ================= FP4: broadcast l4p, concat l3p ========================
    {
        long long M4 = (long long)B * 128;
        pn2_copy_cols<<<cdiv(M4 * 512, 256), 256, 0, stream>>>(l3p, 512, bufC, 1536, 0, M4 * 512);
        pn2_bcast_cols<<<cdiv(M4 * 1024, 256), 256, 0, stream>>>(l4p, 1024, bufC, 1536, 512, 128, M4 * 1024);
        float* y = run_mlp(bufC, (int)M4, fp4, 2); // -> bufB [1024,512]
        pn2_copy_flat<<<cdiv(M4 * 512, 256), 256, 0, stream>>>(y, l3p2, M4 * 512);
    }

    // ================= FP3: interp(l3 -> l2), concat l2p ====================
    {
        long long M3 = (long long)B * 384;
        pn2_copy_cols<<<cdiv(M3 * 256, 256), 256, 0, stream>>>(l2p, 256, bufC, 768, 0, M3 * 256);
        pn2_interp3<<<cdiv(M3, 256), 256, 0, stream>>>(l2x, 384, l3x, 128, l3p2, 512, bufC, 768, 256, (int)M3);
        float* y = run_mlp(bufC, (int)M3, fp3, 2); // -> bufB [3072,256]
        pn2_copy_flat<<<cdiv(M3 * 256, 256), 256, 0, stream>>>(y, l2p2, M3 * 256);
    }

    // ================= FP2: interp(l2 -> l1), concat l1p ====================
    {
        long long M2 = (long long)B * 1024;
        pn2_copy_cols<<<cdiv(M2 * 128, 256), 256, 0, stream>>>(l1p, 128, bufC, 384, 0, M2 * 128);
        pn2_interp3<<<cdiv(M2, 256), 256, 0, stream>>>(l1x, 1024, l2x, 384, l2p2, 256, bufC, 384, 128, (int)M2);
        float* y = run_mlp(bufC, (int)M2, fp2, 2); // -> bufB [8192,128]
        pn2_copy_flat<<<cdiv(M2 * 128, 256), 256, 0, stream>>>(y, l1p2, M2 * 128);
    }

    // ================= FP1: interp(l1 -> l0), MLP, transpose ================
    {
        long long M1 = (long long)B * N;
        pn2_interp3<<<cdiv(M1, 256), 256, 0, stream>>>(xyz, N, l1x, 1024, l1p2, 128, bufC, 128, 0, (int)M1);
        float* y = run_mlp(bufC, (int)M1, fp1, 3); // -> bufA [65536,128]
        pn2_transpose_out<<<cdiv(M1 * 128, 256), 256, 0, stream>>>(
            y, N, 128, M1 * 128, (float*)d_out);
    }
}
